// SelfAttention_13640816132202
// MI455X (gfx1250) — compile-verified, matrix operand order checked by reading, not run on hardware
//
#include <hip/hip_runtime.h>
#include <hip/hip_bf16.h>

typedef __attribute__((ext_vector_type(16))) _Float16 v16h;
typedef __attribute__((ext_vector_type(8)))  _Float16 v8h;
typedef __attribute__((ext_vector_type(4)))  _Float16 v4h;
typedef __attribute__((ext_vector_type(2)))  _Float16 v2h;
typedef __attribute__((ext_vector_type(8)))  float    v8f;
typedef __attribute__((ext_vector_type(4)))  unsigned int v4u;
typedef __attribute__((ext_vector_type(8)))  int      v8i;
typedef __attribute__((ext_vector_type(4)))  int      v4i;

#define WMMA_F16(a, b, c) \
    __builtin_amdgcn_wmma_f32_16x16x32_f16(false, (a), false, (b), (short)0, (c), false, false)

// ---- Tensor Data Mover availability (arity differs between toolchains) ----
#if __has_builtin(__builtin_amdgcn_tensor_load_to_lds) && \
    __has_builtin(__builtin_amdgcn_s_wait_tensorcnt)
#define HAVE_TDM 1
#if defined(__clang_major__) && __clang_major__ >= 23
#define TDM_LOAD(g0, g1, g2, g3) \
    __builtin_amdgcn_tensor_load_to_lds((g0), (g1), (g2), (g3), \
                                        (v8i){0, 0, 0, 0, 0, 0, 0, 0}, 0)
#else
#define TDM_LOAD(g0, g1, g2, g3) \
    __builtin_amdgcn_tensor_load_to_lds((g0), (g1), (g2), (g3), 0)
#endif
#else
#define HAVE_TDM 0
#endif

// ---------------------------------------------------------------------------
// Unified fragment load (wave32 16x16x32 f16): two ds_load_b128 per lane.
//   A 16x32 row-major  : element (m,k) at base + m*ld + k,  lane = row m
//   B 32x16 N-major    : element (k,n) at base + n*ld + k,  lane = col n
// ---------------------------------------------------------------------------
__device__ __forceinline__ v16h frag_ld(const _Float16* base, int ld, int lane) {
    const _Float16* p = base + (lane & 15) * ld + ((lane >> 4) << 3);
    v8h lo = *(const v8h*)p;
    v8h hi = *(const v8h*)(p + 16);
    return __builtin_shufflevector(lo, hi, 0, 1, 2, 3, 4, 5, 6, 7,
                                            8, 9, 10, 11, 12, 13, 14, 15);
}

// ---------------------------------------------------------------------------
// 16-lane (DPP16 row) butterfly reductions on the VALU: no DS, no waits.
// ---------------------------------------------------------------------------
template <int CTRL>
__device__ __forceinline__ float dpp_mv(float v) {
    return __int_as_float(
        __builtin_amdgcn_update_dpp(0, __float_as_int(v), CTRL, 0xf, 0xf, true));
}
__device__ __forceinline__ float row16_max(float v) {
    v = fmaxf(v, dpp_mv<0x128>(v));
    v = fmaxf(v, dpp_mv<0x124>(v));
    v = fmaxf(v, dpp_mv<0x122>(v));
    v = fmaxf(v, dpp_mv<0x121>(v));
    return v;
}
__device__ __forceinline__ float row16_sum(float v) {
    v += dpp_mv<0x128>(v);
    v += dpp_mv<0x124>(v);
    v += dpp_mv<0x122>(v);
    v += dpp_mv<0x121>(v);
    return v;
}

// ---------------------------------------------------------------------------
// Kernel 1: GroupNorm over (16 channels x 1024 spatial), f32 -> f16 xn.
// ---------------------------------------------------------------------------
__global__ __launch_bounds__(256) void gn_kernel(
        const float* __restrict__ x, const float* __restrict__ w,
        const float* __restrict__ bgn, _Float16* __restrict__ xn) {
    const int n   = blockIdx.x >> 5;
    const int g   = blockIdx.x & 31;
    const int tid = threadIdx.x;
    const size_t base = ((size_t)n * 512 + (size_t)g * 16) * 1024;
    const float4* x4 = (const float4*)(x + base);

    float s = 0.f, ss = 0.f;
    for (int i = tid; i < 4096; i += 256) {
        float4 v = x4[i];
        s  += v.x + v.y + v.z + v.w;
        ss += v.x * v.x + v.y * v.y + v.z * v.z + v.w * v.w;
    }
    __shared__ float red0[256], red1[256];
    red0[tid] = s; red1[tid] = ss;
    __syncthreads();
    for (int off = 128; off > 0; off >>= 1) {
        if (tid < off) { red0[tid] += red0[tid + off]; red1[tid] += red1[tid + off]; }
        __syncthreads();
    }
    const float mean = red0[0] * (1.f / 16384.f);
    const float var  = red1[0] * (1.f / 16384.f) - mean * mean;
    const float inv  = rsqrtf(var + 1e-5f);

    for (int i = tid; i < 4096; i += 256) {
        int c = g * 16 + ((i * 4) >> 10);
        float sc = inv * w[c], sh = bgn[c] - mean * sc;
        float4 v = x4[i];
        v4h h;
        h[0] = (_Float16)(v.x * sc + sh);
        h[1] = (_Float16)(v.y * sc + sh);
        h[2] = (_Float16)(v.z * sc + sh);
        h[3] = (_Float16)(v.w * sc + sh);
        *(v4h*)&xn[base + i * 4] = h;
    }
}

// ---------------------------------------------------------------------------
// Kernel 2/4: batched GEMM  out[n] = A(f32 [M,512]) * B(f16 [n][512][1024]) + bias.
// Double-buffered LDS slabs; one barrier per K-step.
// ---------------------------------------------------------------------------
template <bool OUT_F32>
__global__ __launch_bounds__(128, 1) void gemm_kernel(
        const float* __restrict__ A, const _Float16* __restrict__ B,
        const float* __restrict__ bias, void* __restrict__ out, int M) {
    const int lt = blockIdx.x, mt = blockIdx.y, nb = blockIdx.z;
    const int tid  = threadIdx.x;
    const int wave = tid >> 5, lane = tid & 31;
    const int mh = wave >> 1, nh = wave & 1;

    __shared__ alignas(16) _Float16 As [2][64][40];   // [m][k]
    __shared__ alignas(16) _Float16 Bsn[2][64][40];   // [n=l][k] (N-major slab)

    const float*    Ag = A + (size_t)(mt * 64) * 512;
    const _Float16* Bg = B + (size_t)nb * 512 * 1024 + lt * 64;

    auto stage = [&](int k0, int buf) {
#pragma unroll
        for (int i = 0; i < 8; ++i) {              // A slab 64x32: float2 -> v2h
            int idx = tid + i * 128;
            int r = idx >> 4, cp = (idx & 15) << 1;
            float2 f = *(const float2*)&Ag[(size_t)r * 512 + k0 + cp];
            v2h h; h[0] = (_Float16)f.x; h[1] = (_Float16)f.y;
            *(v2h*)&As[buf][r][cp] = h;
        }
#pragma unroll
        for (int i = 0; i < 8; ++i) {              // B slab 32x64 -> N-major
            int idx = tid + i * 128;
            int l = idx & 63, rp = (idx >> 6) << 1;
            v2h h;
            h[0] = Bg[(size_t)(k0 + rp)     * 1024 + l];
            h[1] = Bg[(size_t)(k0 + rp + 1) * 1024 + l];
            *(v2h*)&Bsn[buf][l][rp] = h;
        }
    };

    v8f acc[2][2] = {};
    stage(0, 0);
    __syncthreads();

    for (int ks = 0; ks < 16; ++ks) {
        const int buf = ks & 1;
        if (ks + 1 < 16) stage((ks + 1) * 32, buf ^ 1);

        v16h a0 = frag_ld(&As [buf][mh * 32     ][0], 40, lane);
        v16h a1 = frag_ld(&As [buf][mh * 32 + 16][0], 40, lane);
        v16h b0 = frag_ld(&Bsn[buf][nh * 32     ][0], 40, lane);
        v16h b1 = frag_ld(&Bsn[buf][nh * 32 + 16][0], 40, lane);
        acc[0][0] = WMMA_F16(a0, b0, acc[0][0]);
        acc[0][1] = WMMA_F16(a0, b1, acc[0][1]);
        acc[1][0] = WMMA_F16(a1, b0, acc[1][0]);
        acc[1][1] = WMMA_F16(a1, b1, acc[1][1]);
        __syncthreads();
    }

    const int ncol = lane & 15, half = lane >> 4;
#pragma unroll
    for (int ti = 0; ti < 2; ++ti)
#pragma unroll
    for (int tj = 0; tj < 2; ++tj)
#pragma unroll
        for (int r = 0; r < 8; ++r) {
            int mrow = mt * 64 + mh * 32 + ti * 16 + r + 8 * half;
            int col  = lt * 64 + nh * 32 + tj * 16 + ncol;
            float v  = acc[ti][tj][r] + bias[mrow];
            size_t off = ((size_t)nb * M + mrow) * 1024 + col;
            if constexpr (OUT_F32) ((float*)out)[off] = v;
            else                   ((_Float16*)out)[off] = (_Float16)v;
        }
}

// ---------------------------------------------------------------------------
// Kernel 3: fused flash attention. V tile staged by the Tensor Data Mover
// (2D D#, LDS row padding 128B+16B reproduces the 72-half pitch); K^T staged
// manually (TDM cannot transpose); DPP16 softmax; WMMA for both GEMMs.
// ---------------------------------------------------------------------------
__global__ __launch_bounds__(128, 1) void attn_kernel(
        const _Float16* __restrict__ qkv, _Float16* __restrict__ w2) {
    const int b    = blockIdx.x;          // n*8 + head
    const int ib   = blockIdx.y;          // query block
    const int n    = b >> 3, head = b & 7;
    const int tid  = threadIdx.x;
    const int wave = tid >> 5, lane = tid & 31;
    const int ncol = lane & 15, half = lane >> 4;

    const _Float16* qg  = qkv + ((size_t)n * 1536 + head * 64) * 1024 + ib * 64;
    const _Float16* kgp = qkv + ((size_t)n * 1536 + 512  + head * 64) * 1024;
    const _Float16* vgp = qkv + ((size_t)n * 1536 + 1024 + head * 64) * 1024;

    __shared__ alignas(16) _Float16 Qs [64][72];     // [i][c]  A-native
    __shared__ alignas(16) _Float16 Ksn[64][72];     // [j][c]  B-native (N-major)
    __shared__ alignas(16) _Float16 Vs [64][72];     // [c][j]  B-native for P*V^T
    __shared__ alignas(16) _Float16 Ps [4][16][72];  // per-wave P tile

    const float scale = 0.125f;           // 1/sqrt(64)
#pragma unroll
    for (int i = 0; i < 16; ++i) {        // stage Q^T (scaled), channel pairs
        int idx = tid + i * 128;
        int q = idx & 63, cp = (idx >> 6) << 1;
        v2h h;
        h[0] = (_Float16)((float)qg[(size_t)cp       * 1024 + q] * scale);
        h[1] = (_Float16)((float)qg[(size_t)(cp + 1) * 1024 + q] * scale);
        *(v2h*)&Qs[q][cp] = h;
    }
    __syncthreads();

    // loop-invariant Q fragments (rows wave*16..+15, K-chunks 0 and 32)
    v16h qf0 = frag_ld(&Qs[wave * 16][0],  72, lane);
    v16h qf1 = frag_ld(&Qs[wave * 16][32], 72, lane);

    float mstate[8], lstate[8];
    v8f oacc[4] = {};
#pragma unroll
    for (int r = 0; r < 8; ++r) { mstate[r] = -1e30f; lstate[r] = 0.f; }

#if HAVE_TDM
    const unsigned lds_v = (unsigned)(size_t)(void*)&Vs[0][0];  // LDS byte offset
#endif

    for (int jb = 0; jb < 16; ++jb, kgp += 64, vgp += 64) {
#if HAVE_TDM
        if (wave == 0) {   // async V tile: 64x64 f16, LDS pad 16B per 128B row
            unsigned long long ga = (unsigned long long)(size_t)vgp;
            v4u g0;
            g0[0] = 1u;                                   // count=1 (user D#)
            g0[1] = lds_v;                                // lds_addr
            g0[2] = (unsigned)ga;                         // global_addr[31:0]
            g0[3] = (unsigned)(ga >> 32) | (2u << 30);    // ga[56:32] | type=2
            v8i g1;
            g1[0] = (1 << 16) | (1 << 20) | (4 << 22) | (3 << 25);
            //        data=2B   pad_en      ivl=32dw     amt=4dw
            g1[1] = (int)(1024u << 16);                   // tensor_dim0 = 1024
            g1[2] = (int)(64u   << 16);                   // tensor_dim1 = 64
            g1[3] = (int)(64u   << 16);                   // tile_dim0   = 64
            g1[4] = 64;                                   // tile_dim1=64, tile_dim2=0
            g1[5] = 1024;                                 // tensor_dim0_stride
            g1[6] = 0;
            g1[7] = 0;
            v4i gz = {0, 0, 0, 0};
            TDM_LOAD(g0, g1, gz, gz);
        }
#endif
        {   // prefetch next K block (global_prefetch_b8)
            int c = tid & 63;
            __builtin_prefetch(&kgp[(size_t)c * 1024 + 64], 0, 1);
        }
#pragma unroll
        for (int i = 0; i < 16; ++i) {    // stage K^T (N-major) - overlaps TDM
            int idx = tid + i * 128;
            int j = idx & 63, cp = (idx >> 6) << 1;
            v2h h;
            h[0] = kgp[(size_t)cp       * 1024 + j];
            h[1] = kgp[(size_t)(cp + 1) * 1024 + j];
            *(v2h*)&Ksn[j][cp] = h;
        }
#if HAVE_TDM
        if (wave == 0) __builtin_amdgcn_s_wait_tensorcnt(0);
#else
#pragma unroll
        for (int i = 0; i < 8; ++i) {     // fallback: manual V staging
            int idx = tid + i * 128;
            int c = idx >> 4, j4 = (idx & 15) << 2;
            *(v4h*)&Vs[c][j4] = *(const v4h*)&vgp[(size_t)c * 1024 + j4];
        }
#endif
        __syncthreads();

        // S = (scaled Q)^T K : 16 rows x 64 cols per wave
        v8f s[4] = {};
#pragma unroll
        for (int t = 0; t < 4; ++t) {
            v16h bf0 = frag_ld(&Ksn[t * 16][0],  72, lane);
            v16h bf1 = frag_ld(&Ksn[t * 16][32], 72, lane);
            s[t] = WMMA_F16(qf0, bf0, s[t]);
            s[t] = WMMA_F16(qf1, bf1, s[t]);
        }

        // fused online softmax per row; reductions on DPP16 butterflies
#pragma unroll
        for (int r = 0; r < 8; ++r) {
            float rowm = fmaxf(fmaxf(s[0][r], s[1][r]), fmaxf(s[2][r], s[3][r]));
            rowm = row16_max(rowm);
            float mnew  = fmaxf(mstate[r], rowm);
            float alpha = __expf(mstate[r] - mnew);
            mstate[r] = mnew;
            float p0 = __expf(s[0][r] - mnew);
            float p1 = __expf(s[1][r] - mnew);
            float p2 = __expf(s[2][r] - mnew);
            float p3 = __expf(s[3][r] - mnew);
            s[0][r] = p0; s[1][r] = p1; s[2][r] = p2; s[3][r] = p3;
            float rsum = row16_sum(p0 + p1 + p2 + p3);
            lstate[r] = lstate[r] * alpha + rsum;
            oacc[0][r] *= alpha; oacc[1][r] *= alpha;
            oacc[2][r] *= alpha; oacc[3][r] *= alpha;
        }

        // C-layout -> A-layout via per-wave LDS (DS in-order within a wave)
#pragma unroll
        for (int t = 0; t < 4; ++t)
#pragma unroll
            for (int r = 0; r < 8; ++r)
                Ps[wave][r + 8 * half][t * 16 + ncol] = (_Float16)s[t][r];

        // O += P * V^T
#pragma unroll
        for (int kk = 0; kk < 64; kk += 32) {
            v16h af = frag_ld(&Ps[wave][0][kk], 72, lane);
#pragma unroll
            for (int t = 0; t < 4; ++t) {
                v16h bf = frag_ld(&Vs[t * 16][kk], 72, lane);
                oacc[t] = WMMA_F16(af, bf, oacc[t]);
            }
        }
        __syncthreads();
    }

#pragma unroll
    for (int r = 0; r < 8; ++r) {
        float inv = 1.f / lstate[r];
#pragma unroll
        for (int t = 0; t < 4; ++t) oacc[t][r] *= inv;
    }

    // torch-faithful layout: per n, flat index head*65536 + i*64 + c
    _Float16* w2n = w2 + (size_t)n * 524288 + (size_t)head * 65536;
#pragma unroll
    for (int t = 0; t < 4; ++t)
#pragma unroll
        for (int r = 0; r < 8; ++r) {
            int i  = ib * 64 + wave * 16 + r + 8 * half;
            int cc = t * 16 + ncol;
            w2n[(size_t)i * 64 + cc] = (_Float16)oacc[t][r];
        }
}

// ---------------------------------------------------------------------------
extern "C" void kernel_launch(void* const* d_in, const int* in_sizes, int n_in,
                              void* d_out, int out_size, void* d_ws, size_t ws_size,
                              hipStream_t stream) {
    const float* x    = (const float*)d_in[0];
    const float* gnw  = (const float*)d_in[1];
    const float* gnb  = (const float*)d_in[2];
    const float* qkvw = (const float*)d_in[3];
    const float* qkvb = (const float*)d_in[4];
    const float* outw = (const float*)d_in[5];
    const float* outb = (const float*)d_in[6];

    // workspace (f16): xn 8MB | qkv 24MB | weighted 8MB  (fits in 192MB L2)
    _Float16* xn   = (_Float16*)d_ws;
    _Float16* qkvh = (_Float16*)((char*)d_ws + (size_t)8  * 1024 * 1024);
    _Float16* w2   = (_Float16*)((char*)d_ws + (size_t)32 * 1024 * 1024);

    gn_kernel<<<dim3(256), 256, 0, stream>>>(x, gnw, gnb, xn);
    gemm_kernel<false><<<dim3(16, 24, 8), 128, 0, stream>>>(qkvw, xn, qkvb, (void*)qkvh, 1536);
    attn_kernel<<<dim3(64, 16), 128, 0, stream>>>(qkvh, w2);
    gemm_kernel<true ><<<dim3(16,  8, 8), 128, 0, stream>>>(outw, w2, outb, d_out, 512);
}
